// GraphNetBlock_13219909337176
// MI455X (gfx1250) — matrix operand order repaired and checked by reading, not compiled, not run on hardware
//
#include <hip/hip_runtime.h>

#define TILE_ROWS 32
#define LDA   392   /* padded bf16 row stride for 384-wide A stage */
#define LDH1  136   /* padded bf16 row stride for 128-wide h1 */
#define LDH2  132   /* padded f32  row stride for 128-wide h2 */
#define SMEM_BYTES (TILE_ROWS*LDA*2 + TILE_ROWS*LDH1*2)

typedef __attribute__((ext_vector_type(16))) __bf16        v16bf;
typedef __attribute__((ext_vector_type(8)))  float         v8f;
typedef __attribute__((ext_vector_type(8)))  unsigned int  v8u;

__device__ __forceinline__ unsigned short f2bf(float f) {
  unsigned int u = __float_as_uint(f);
  u = (u + 0x7FFFu + ((u >> 16) & 1u)) >> 16;   // round-to-nearest-even
  return (unsigned short)u;
}

// A-fragment (16x32 bf16, MxK) from LDS, per CDNA5 ISA layout:
// lane = half*16 + m ; VGPR v holds K pair: K = (v&4?16:0) + (v&3)*2 + half*8
__device__ __forceinline__ v16bf load_a_frag(const unsigned short* sA, int lda,
                                             int r0, int kk) {
  int lane = threadIdx.x & 31;
  int m    = lane & 15;
  int half = lane >> 4;
  v8u u;
#pragma unroll
  for (int v = 0; v < 8; ++v) {
    int kb = ((v & 4) ? 16 : 0) + (v & 3) * 2 + half * 8;
    u[v] = *(const unsigned int*)(&sA[(r0 + m) * lda + kk + kb]);
  }
  return __builtin_bit_cast(v16bf, u);
}

// B-fragment (32x16 bf16, KxN) from pre-packed weights: one contiguous
// 32B load per lane. Packed order: (((kt*8 + nt)*32 + lane)*8 + v)
__device__ __forceinline__ v16bf load_b_frag(const unsigned int* __restrict__ Wp,
                                             int kt, int nt) {
  int lane = threadIdx.x & 31;
  const unsigned int* p = Wp + (((kt * 8 + nt) * 32 + lane) * 8);
  v8u u;
#pragma unroll
  for (int v = 0; v < 8; ++v) u[v] = p[v];
  return __builtin_bit_cast(v16bf, u);
}

// Pack fp32 KxN weight matrix into per-lane B-fragment order (bf16 pairs).
// Lane half h, elem v -> K = kt*32 + (v&4?16:0) + (v&3)*2 + h*8 (+elem j), N = nt*16 + lane%16.
__global__ void pack_w(const float* __restrict__ W, unsigned int* __restrict__ out,
                       int K, int N) {
  int idx = blockIdx.x * blockDim.x + threadIdx.x;
  int NT = N / 16;
  int total = (K / 32) * NT * 32 * 8;
  if (idx >= total) return;
  int v    = idx & 7;
  int lane = (idx >> 3) & 31;
  int t    = idx >> 8;
  int nt   = t % NT;
  int kt   = t / NT;
  int half = lane >> 4;
  int n    = nt * 16 + (lane & 15);
  int k    = kt * 32 + ((v & 4) ? 16 : 0) + (v & 3) * 2 + half * 8;
  unsigned int lo = f2bf(W[(long)k * N + n]);
  unsigned int hi = f2bf(W[(long)(k + 1) * N + n]);
  out[idx] = lo | (hi << 16);
}

// Two-layer MLP on a staged 32x384 bf16 A tile. Leaves h2 (=x@W1 relu @W2 + b2)
// in sH2 (fp32, 32 x LDH2) and ends with a barrier.
__device__ __forceinline__ void mlp_core(const unsigned short* sA,
                                         unsigned short* sH1, float* sH2,
                                         const unsigned int* __restrict__ W1p,
                                         const float* __restrict__ b1,
                                         const unsigned int* __restrict__ W2p,
                                         const float* __restrict__ b2) {
  int wave = threadIdx.x >> 5;      // 0..7 -> N-tile
  int lane = threadIdx.x & 31;
  int nloc = lane & 15;
  int half = lane >> 4;
  int n0   = wave * 16;

  // ---- layer 1: [32x384] @ [384x128], K = 12 steps of 32 ----
  v8f acc0 = {};
  v8f acc1 = {};
#pragma unroll
  for (int kt = 0; kt < 12; ++kt) {
    v16bf bf = load_b_frag(W1p, kt, wave);
    v16bf a0 = load_a_frag(sA, LDA, 0, kt * 32);
    acc0 = __builtin_amdgcn_wmma_f32_16x16x32_bf16(false, a0, false, bf,
                                                   (short)0, acc0, false, false);
    v16bf a1 = load_a_frag(sA, LDA, 16, kt * 32);
    acc1 = __builtin_amdgcn_wmma_f32_16x16x32_bf16(false, a1, false, bf,
                                                   (short)0, acc1, false, false);
  }
  float bias1 = b1[n0 + nloc];
#pragma unroll
  for (int v = 0; v < 8; ++v) {     // C/D layout: lane->N, VGPR v -> M = v + 8*half
    float x0 = acc0[v] + bias1; x0 = x0 > 0.f ? x0 : 0.f;
    float x1 = acc1[v] + bias1; x1 = x1 > 0.f ? x1 : 0.f;
    int r = v + 8 * half;
    sH1[r * LDH1 + n0 + nloc]        = f2bf(x0);
    sH1[(16 + r) * LDH1 + n0 + nloc] = f2bf(x1);
  }
  __syncthreads();   // all waves done with sA reads + h1 writes

  // ---- layer 2: [32x128] @ [128x128], K = 4 steps of 32 ----
  v8f d0 = {};
  v8f d1 = {};
#pragma unroll
  for (int kt = 0; kt < 4; ++kt) {
    v16bf bf = load_b_frag(W2p, kt, wave);
    v16bf a0 = load_a_frag(sH1, LDH1, 0, kt * 32);
    d0 = __builtin_amdgcn_wmma_f32_16x16x32_bf16(false, a0, false, bf,
                                                 (short)0, d0, false, false);
    v16bf a1 = load_a_frag(sH1, LDH1, 16, kt * 32);
    d1 = __builtin_amdgcn_wmma_f32_16x16x32_bf16(false, a1, false, bf,
                                                 (short)0, d1, false, false);
  }
  float bias2 = b2[n0 + nloc];
#pragma unroll
  for (int v = 0; v < 8; ++v) {
    int r = v + 8 * half;
    sH2[r * LDH2 + n0 + nloc]        = d0[v] + bias2;   // sH2 overlaps sA: safe, sA dead
    sH2[(16 + r) * LDH2 + n0 + nloc] = d1[v] + bias2;
  }
  __syncthreads();
}

// ---- fused edge update: gather -> MLP -> LN -> (out = ln + efeat), scatter ln ----
__global__ __launch_bounds__(256) void edge_kernel(
    const float* __restrict__ nf,
    const int* __restrict__ senders, const int* __restrict__ receivers,
    const float* __restrict__ efeat,
    const unsigned int* __restrict__ W1p, const float* __restrict__ b1,
    const unsigned int* __restrict__ W2p, const float* __restrict__ b2,
    const float* __restrict__ g, const float* __restrict__ beta,
    float* __restrict__ out_e, float* __restrict__ agg, int E) {
  __shared__ __align__(16) char smem[SMEM_BYTES];
  unsigned short* sA  = (unsigned short*)smem;
  unsigned short* sH1 = (unsigned short*)(smem + TILE_ROWS * LDA * 2);
  float*          sH2 = (float*)smem;

  long e0 = (long)blockIdx.x * TILE_ROWS;
  // stage concat[nf[s], nf[r], ef] as bf16, coalesced across threads
  for (int i = threadIdx.x; i < TILE_ROWS * 384; i += 256) {
    int row = i / 384;
    int col = i - row * 384;
    long e = e0 + row;
    float x = 0.f;
    if (e < E) {
      if (col < 128)      x = nf[(long)senders[e] * 128 + col];
      else if (col < 256) x = nf[(long)receivers[e] * 128 + (col - 128)];
      else                x = efeat[e * 128 + (col - 256)];
    }
    sA[row * LDA + col] = f2bf(x);
  }
  __syncthreads();

  mlp_core(sA, sH1, sH2, W1p, b1, W2p, b2);

  // LayerNorm: 8 lanes per row, 16 cols per lane
  int row = threadIdx.x >> 3;
  int c0  = (threadIdx.x & 7) * 16;
  float vals[16];
  float s = 0.f, ss = 0.f;
#pragma unroll
  for (int j = 0; j < 16; ++j) {
    float x = sH2[row * LDH2 + c0 + j];
    vals[j] = x; s += x; ss += x * x;
  }
#pragma unroll
  for (int off = 4; off >= 1; off >>= 1) {
    s  += __shfl_xor(s,  off, 8);
    ss += __shfl_xor(ss, off, 8);
  }
  float mu   = s * (1.f / 128.f);
  float var  = ss * (1.f / 128.f) - mu * mu;
  float rstd = rsqrtf(var + 1e-5f);

  long e = e0 + row;
  if (e < E) {
    int rcv = receivers[e];
    const float* ef = efeat + e * 128;
    float* op = out_e + e * 128;
    float* ag = agg + (long)rcv * 128;
#pragma unroll
    for (int j = 0; j < 16; ++j) {
      int c = c0 + j;
      float ln = g[c] * (vals[j] - mu) * rstd + beta[c];
      op[c] = ln + ef[c];                 // residual output
      unsafeAtomicAdd(&ag[c], ln);        // global_atomic_add_f32 into L2-resident agg
    }
  }
}

// ---- fused node update: concat[nf, agg_mesh, agg_world] -> MLP -> LN -> +nf ----
__global__ __launch_bounds__(256) void node_kernel(
    const float* __restrict__ nf,
    const float* __restrict__ aggm, const float* __restrict__ aggw,
    const unsigned int* __restrict__ W1p, const float* __restrict__ b1,
    const unsigned int* __restrict__ W2p, const float* __restrict__ b2,
    const float* __restrict__ g, const float* __restrict__ beta,
    float* __restrict__ out_n, int N) {
  __shared__ __align__(16) char smem[SMEM_BYTES];
  unsigned short* sA  = (unsigned short*)smem;
  unsigned short* sH1 = (unsigned short*)(smem + TILE_ROWS * LDA * 2);
  float*          sH2 = (float*)smem;

  long n0g = (long)blockIdx.x * TILE_ROWS;
  for (int i = threadIdx.x; i < TILE_ROWS * 384; i += 256) {
    int row = i / 384;
    int col = i - row * 384;
    long n = n0g + row;
    float x = 0.f;
    if (n < N) {
      if (col < 128)      x = nf[n * 128 + col];
      else if (col < 256) x = aggm[n * 128 + (col - 128)];
      else                x = aggw[n * 128 + (col - 256)];
    }
    sA[row * LDA + col] = f2bf(x);
  }
  __syncthreads();

  mlp_core(sA, sH1, sH2, W1p, b1, W2p, b2);

  int row = threadIdx.x >> 3;
  int c0  = (threadIdx.x & 7) * 16;
  float vals[16];
  float s = 0.f, ss = 0.f;
#pragma unroll
  for (int j = 0; j < 16; ++j) {
    float x = sH2[row * LDH2 + c0 + j];
    vals[j] = x; s += x; ss += x * x;
  }
#pragma unroll
  for (int off = 4; off >= 1; off >>= 1) {
    s  += __shfl_xor(s,  off, 8);
    ss += __shfl_xor(ss, off, 8);
  }
  float mu   = s * (1.f / 128.f);
  float var  = ss * (1.f / 128.f) - mu * mu;
  float rstd = rsqrtf(var + 1e-5f);

  long n = n0g + row;
  if (n < N) {
    const float* nfr = nf + n * 128;
    float* op = out_n + n * 128;
#pragma unroll
    for (int j = 0; j < 16; ++j) {
      int c = c0 + j;
      op[c] = g[c] * (vals[j] - mu) * rstd + beta[c] + nfr[c];
    }
  }
}

extern "C" void kernel_launch(void* const* d_in, const int* in_sizes, int n_in,
                              void* d_out, int out_size, void* d_ws, size_t ws_size,
                              hipStream_t stream) {
  const float* nf    = (const float*)d_in[0];
  const int*   ms    = (const int*)d_in[1];
  const int*   mr    = (const int*)d_in[2];
  const float* mf    = (const float*)d_in[3];
  const int*   wsnd  = (const int*)d_in[4];
  const int*   wrcv  = (const int*)d_in[5];
  const float* wf    = (const float*)d_in[6];
  const float* meW1  = (const float*)d_in[7];
  const float* meb1  = (const float*)d_in[8];
  const float* meW2  = (const float*)d_in[9];
  const float* meb2  = (const float*)d_in[10];
  const float* meg   = (const float*)d_in[11];
  const float* mebt  = (const float*)d_in[12];
  const float* weW1  = (const float*)d_in[13];
  const float* web1  = (const float*)d_in[14];
  const float* weW2  = (const float*)d_in[15];
  const float* web2  = (const float*)d_in[16];
  const float* weg   = (const float*)d_in[17];
  const float* webt  = (const float*)d_in[18];
  const float* nmW1  = (const float*)d_in[19];
  const float* nmb1  = (const float*)d_in[20];
  const float* nmW2  = (const float*)d_in[21];
  const float* nmb2  = (const float*)d_in[22];
  const float* nmg   = (const float*)d_in[23];
  const float* nmbt  = (const float*)d_in[24];

  int N  = in_sizes[0] / 128;
  int EM = in_sizes[1];
  int EW = in_sizes[4];

  float* out_nodes = (float*)d_out;
  float* out_mesh  = out_nodes + (long)N * 128;
  float* out_world = out_mesh + (long)EM * 128;

  // workspace: [agg_mesh f32][agg_world f32][6 packed weight blobs]
  float* aggm = (float*)d_ws;
  float* aggw = aggm + (long)N * 128;
  unsigned int* wp = (unsigned int*)(aggw + (long)N * 128);
  const int W1P = (384 / 32) * 8 * 32 * 8;   // 24576 uints
  const int W2P = (128 / 32) * 8 * 32 * 8;   // 8192 uints
  unsigned int* meW1p = wp;
  unsigned int* meW2p = meW1p + W1P;
  unsigned int* weW1p = meW2p + W2P;
  unsigned int* weW2p = weW1p + W1P;
  unsigned int* nmW1p = weW2p + W2P;
  unsigned int* nmW2p = nmW1p + W1P;

  hipMemsetAsync(aggm, 0, (size_t)N * 128 * 2 * sizeof(float), stream);

  pack_w<<<(W1P + 255) / 256, 256, 0, stream>>>(meW1, meW1p, 384, 128);
  pack_w<<<(W2P + 255) / 256, 256, 0, stream>>>(meW2, meW2p, 128, 128);
  pack_w<<<(W1P + 255) / 256, 256, 0, stream>>>(weW1, weW1p, 384, 128);
  pack_w<<<(W2P + 255) / 256, 256, 0, stream>>>(weW2, weW2p, 128, 128);
  pack_w<<<(W1P + 255) / 256, 256, 0, stream>>>(nmW1, nmW1p, 384, 128);
  pack_w<<<(W2P + 255) / 256, 256, 0, stream>>>(nmW2, nmW2p, 128, 128);

  edge_kernel<<<(EM + TILE_ROWS - 1) / TILE_ROWS, 256, 0, stream>>>(
      nf, ms, mr, mf, meW1p, meb1, meW2p, meb2, meg, mebt, out_mesh, aggm, EM);
  edge_kernel<<<(EW + TILE_ROWS - 1) / TILE_ROWS, 256, 0, stream>>>(
      nf, wsnd, wrcv, wf, weW1p, web1, weW2p, web2, weg, webt, out_world, aggw, EW);
  node_kernel<<<(N + TILE_ROWS - 1) / TILE_ROWS, 256, 0, stream>>>(
      nf, aggm, aggw, nmW1p, nmb1, nmW2p, nmb2, nmg, nmbt, out_nodes, N);
}
